// StructureModuleRNA_63763084476998
// MI455X (gfx1250) — matrix operand order repaired and checked by reading, not compiled, not run on hardware
//
#include <hip/hip_runtime.h>
#include <math.h>
#include <stdint.h>

// ---------------------------------------------------------------------------
// Types & helpers for CDNA5 (gfx1250) wave32 WMMA bf16
// ---------------------------------------------------------------------------
typedef unsigned short bf16_t;
typedef __attribute__((ext_vector_type(16))) __bf16 bf16x16;
typedef __attribute__((ext_vector_type(8)))  float  f32x8;
typedef __attribute__((ext_vector_type(4)))  unsigned short u16x4;
typedef __attribute__((ext_vector_type(8)))  unsigned short u16x8;

union FragAB { unsigned short u[16]; u16x8 h[2]; bf16x16 v; };
union FragCD { float f[8]; f32x8 v; };

__device__ __forceinline__ bf16_t f2bf(float x) {
  union { float f; unsigned int u; } c; c.f = x;
  unsigned int u = c.u;
  unsigned int r = (u + 0x7FFFu + ((u >> 16) & 1u)) >> 16;
  return (bf16_t)r;
}
__device__ __forceinline__ float bf2f(bf16_t b) {
  union { unsigned int u; float f; } c; c.u = ((unsigned int)b) << 16; return c.f;
}

// A-fragment (16x32 bf16, row-major MxK) K index for element e, lane-half h
__device__ __forceinline__ int a_k_of(int e, int half) {
  return half * 16 + e;
}
// B-fragment (32x16 bf16): lane (n=l&15, half=l>>4), element e -> k = half*16+e

__device__ __forceinline__ f32x8 wmma_bf16(const FragAB& a, const FragAB& b, f32x8 c) {
  return __builtin_amdgcn_wmma_f32_16x16x32_bf16(
      false, a.v, false, b.v, (short)0, c, false, false);
}

// CDNA5 async global->LDS copy (16B per lane), ASYNCcnt-tracked.
__device__ __forceinline__ void async_ld_b128(unsigned lds_off, unsigned long long gaddr) {
  asm volatile("global_load_async_to_lds_b128 %0, %1, off"
               :: "v"(lds_off), "v"(gaddr) : "memory");
}
__device__ __forceinline__ void wait_async0() {
  asm volatile("s_wait_asynccnt 0x0" ::: "memory");
}

// ---------------------------------------------------------------------------
// Model constants
// ---------------------------------------------------------------------------
#define NRES 512
#define CS   384
#define CZ   128
#define NH   12
#define CI   16
#define PQ   4
#define PV   8
#define FEAT 2112   // H*(CI + CZ + PV*4)
#define WLC  0.5773502691896258f      // sqrt(1/3)
#define WCC  0.23570226039551584f     // sqrt(2/(9*PQ)) = sqrt(1/18)

#define GT_RELU_IN  1
#define GT_RELU_OUT 2
#define GT_RESID    4

// ---------------------------------------------------------------------------
// Weight packing: W[K][N] f32 -> Wt[Npad][K] bf16 (transposed, zero-padded)
// ---------------------------------------------------------------------------
__global__ void pack_weight(const float* __restrict__ W, bf16_t* __restrict__ Wt,
                            int K, int N, int Npad)
{
  int idx = blockIdx.x * blockDim.x + threadIdx.x;
  if (idx >= Npad * K) return;
  int n = idx / K, k = idx % K;
  Wt[idx] = (n < N) ? f2bf(W[(size_t)k * N + n]) : (bf16_t)0;
}

// ---------------------------------------------------------------------------
// Generic WMMA GEMM: out[M,N] = epilogue(A[M,K] @ W[K,N] + bias [+ resid])
// Block = 128 threads (4 waves), tile 64(M) x 64(N), K-steps of 32.
// B tile: async global->LDS DMA of pre-packed bf16 weights (no conversion).
// A tile: f32->bf16 in VGPRs, stored to LDS pre-swizzled into fragment order
//         so every WMMA fragment is two ds_load_b128.
// Requires: M % 64 == 0, K % 32 == 0 (true for every GEMM in this model).
// ---------------------------------------------------------------------------
__device__ __forceinline__ void loadA4(const float* p, float v[4]) {
  float4 t = *(const float4*)p;
  v[0] = t.x; v[1] = t.y; v[2] = t.z; v[3] = t.w;
}
__device__ __forceinline__ void loadA4(const bf16_t* p, float v[4]) {
  u16x4 t = *(const u16x4*)p;
  v[0] = bf2f(t[0]); v[1] = bf2f(t[1]); v[2] = bf2f(t[2]); v[3] = bf2f(t[3]);
}

template <typename TA>
__global__ __launch_bounds__(128)
void gemm_wmma(const TA* __restrict__ A, const bf16_t* __restrict__ Wt,
               const float* __restrict__ bias, const float* resid,
               float* out, int M, int K, int N, int flags)
{
  __shared__ alignas(16) unsigned short As[4 * 32 * 16];  // [sub][lane][elem]
  __shared__ alignas(16) unsigned short Bs[64 * 32];      // [n][k]
  const int m0 = blockIdx.y * 64, n0 = blockIdx.x * 64;
  const int tid = threadIdx.x;
  const int w = tid >> 5, l = tid & 31;
  const int lm = l & 15, half = (l >> 4) & 1;

  FragCD acc[4];
  for (int nt = 0; nt < 4; nt++)
    for (int r = 0; r < 8; r++) acc[nt].f[r] = 0.f;

  for (int kk = 0; kk < K; kk += 32) {
    // ---- async DMA of B tile: 64 rows x 32 bf16 = 4KB = 256 x b128 ----
#pragma unroll
    for (int jj = 0; jj < 2; jj++) {
      int t2 = tid + 128 * jj;
      int r = t2 >> 2, seg = t2 & 3;
      unsigned lds_off = (unsigned)(uintptr_t)&Bs[r * 32 + seg * 8];
      unsigned long long ga =
          (unsigned long long)(uintptr_t)(Wt + (size_t)(n0 + r) * K + kk + seg * 8);
      async_ld_b128(lds_off, ga);
    }
    // ---- A tile: 64 x 32, vectorized loads + swizzled bf16 store ----
#pragma unroll
    for (int ii = 0; ii < 4; ii++) {
      int t4 = tid + 128 * ii;
      int row = t4 >> 3;
      int c4 = (t4 & 7) << 2;
      float v[4];
      loadA4(A + (size_t)(m0 + row) * K + kk + c4, v);
      if (flags & GT_RELU_IN) {
        v[0] = fmaxf(v[0], 0.f); v[1] = fmaxf(v[1], 0.f);
        v[2] = fmaxf(v[2], 0.f); v[3] = fmaxf(v[3], 0.f);
      }
      u16x4 pk;
      pk[0] = f2bf(v[0]); pk[1] = f2bf(v[1]); pk[2] = f2bf(v[2]); pk[3] = f2bf(v[3]);
      int hf = (c4 >> 3) & 1;
      int lane = (row & 15) + 16 * hf;
      int e0 = 2 * (((c4 >> 4) << 2) | ((c4 >> 1) & 3));
      *(u16x4*)&As[((row >> 4) * 32 + lane) * 16 + e0] = pk;
    }
    wait_async0();
    __syncthreads();

    FragAB a;
    {
      const u16x8* ap = (const u16x8*)&As[(w * 32 + l) * 16];
      a.h[0] = ap[0];
      a.h[1] = ap[1];
    }
#pragma unroll
    for (int nt = 0; nt < 4; nt++) {
      FragAB b;
      const u16x8* bp = (const u16x8*)&Bs[(nt * 16 + lm) * 32 + half * 8];
      b.h[0] = bp[0];
      b.h[1] = bp[2];
      acc[nt].v = wmma_bf16(a, b, acc[nt].v);
    }
    __syncthreads();
  }

#pragma unroll
  for (int nt = 0; nt < 4; nt++) {
    int gc = n0 + nt * 16 + lm;
#pragma unroll
    for (int r = 0; r < 8; r++) {
      int gr = m0 + w * 16 + r + 8 * half;
      if (gc < N) {
        float v = acc[nt].f[r];
        if (bias) v += bias[gc];
        if (flags & GT_RESID) v += resid[(size_t)gr * N + gc];
        if (flags & GT_RELU_OUT) v = fmaxf(v, 0.f);
        out[(size_t)gr * N + gc] = v;
      }
    }
  }
}

// ---------------------------------------------------------------------------
// LayerNorm (one row per block). In-place safe.
// ---------------------------------------------------------------------------
__global__ void ln_f32(const float* x, const float* g, const float* b,
                       float* out, int dim)
{
  const int row = blockIdx.x;
  const float* xr = x + (size_t)row * dim;
  float* orow = out + (size_t)row * dim;
  __shared__ float s1[128], s2[128];
  float sum = 0.f, sq = 0.f;
  for (int c = threadIdx.x; c < dim; c += blockDim.x) {
    float v = xr[c]; sum += v; sq += v * v;
  }
  s1[threadIdx.x] = sum; s2[threadIdx.x] = sq;
  __syncthreads();
  for (int off = blockDim.x >> 1; off > 0; off >>= 1) {
    if ((int)threadIdx.x < off) {
      s1[threadIdx.x] += s1[threadIdx.x + off];
      s2[threadIdx.x] += s2[threadIdx.x + off];
    }
    __syncthreads();
  }
  float mu = s1[0] / dim;
  float var = s2[0] / dim - mu * mu;
  float inv = rsqrtf(var + 1e-5f);
  for (int c = threadIdx.x; c < dim; c += blockDim.x)
    orow[c] = (xr[c] - mu) * inv * g[c] + b[c];
}

__global__ void ln_f32_to_bf16(const float* x, const float* g, const float* b,
                               bf16_t* out, int dim)
{
  const int row = blockIdx.x;
  const float* xr = x + (size_t)row * dim;
  bf16_t* orow = out + (size_t)row * dim;
  __shared__ float s1[128], s2[128];
  float sum = 0.f, sq = 0.f;
  for (int c = threadIdx.x; c < dim; c += blockDim.x) {
    float v = xr[c]; sum += v; sq += v * v;
  }
  s1[threadIdx.x] = sum; s2[threadIdx.x] = sq;
  __syncthreads();
  for (int off = blockDim.x >> 1; off > 0; off >>= 1) {
    if ((int)threadIdx.x < off) {
      s1[threadIdx.x] += s1[threadIdx.x + off];
      s2[threadIdx.x] += s2[threadIdx.x + off];
    }
    __syncthreads();
  }
  float mu = s1[0] / dim;
  float var = s2[0] / dim - mu * mu;
  float inv = rsqrtf(var + 1e-5f);
  for (int c = threadIdx.x; c < dim; c += blockDim.x)
    orow[c] = f2bf((xr[c] - mu) * inv * g[c] + b[c]);
}

// ---------------------------------------------------------------------------
// Small elementwise kernels
// ---------------------------------------------------------------------------
__global__ void fill0(float* p, int n)
{
  int i = blockIdx.x * blockDim.x + threadIdx.x;
  if (i < n) p[i] = 0.f;
}

__global__ void addvec(float* dst, const float* src, int n)
{
  int i = blockIdx.x * blockDim.x + threadIdx.x;
  if (i < n) dst[i] += src[i];
}

// Rigid frame from 3 points: Rt[n] = [R row-major (9), t (3)], t scaled.
__global__ void frames_from3(const float* xyz, int ia, int ib, int ic,
                             float tscale, float* Rt)
{
  int n = blockIdx.x * blockDim.x + threadIdx.x;
  if (n >= NRES) return;
  const float* p = xyz + n * 15;
  float ax = p[ia*3+0], ay = p[ia*3+1], az = p[ia*3+2];
  float ox = p[ib*3+0], oy = p[ib*3+1], oz = p[ib*3+2];
  float cx = p[ic*3+0], cy = p[ic*3+1], cz = p[ic*3+2];
  float e0x = ox - ax, e0y = oy - ay, e0z = oz - az;
  float e1x = cx - ox, e1y = cy - oy, e1z = cz - oz;
  float i0 = rsqrtf(e0x*e0x + e0y*e0y + e0z*e0z + 1e-8f);
  e0x *= i0; e0y *= i0; e0z *= i0;
  float d = e1x*e0x + e1y*e0y + e1z*e0z;
  e1x -= d * e0x; e1y -= d * e0y; e1z -= d * e0z;
  float i1 = rsqrtf(e1x*e1x + e1y*e1y + e1z*e1z + 1e-8f);
  e1x *= i1; e1y *= i1; e1z *= i1;
  float e2x = e0y*e1z - e0z*e1y;
  float e2y = e0z*e1x - e0x*e1z;
  float e2z = e0x*e1y - e0y*e1x;
  float* R = Rt + n * 12;
  R[0] = e0x; R[1] = e1x; R[2] = e2x;
  R[3] = e0y; R[4] = e1y; R[5] = e2y;
  R[6] = e0z; R[7] = e1z; R[8] = e2z;
  R[9] = ox * tscale; R[10] = oy * tscale; R[11] = oz * tscale;
}

// glob[n][p] = R[n] @ loc[n][p] + t[n]      (npts points of 3 per residue)
__global__ void to_global(const float* loc, const float* Rt, float* glob, int npts)
{
  int idx = blockIdx.x * blockDim.x + threadIdx.x;
  if (idx >= NRES * npts) return;
  int n = idx / npts, p = idx % npts;
  const float* R = Rt + n * 12;
  const float* l = loc + (size_t)n * npts * 3 + p * 3;
  float x = l[0], y = l[1], z = l[2];
  float* o = glob + (size_t)n * npts * 3 + p * 3;
  o[0] = R[0]*x + R[1]*y + R[2]*z + R[9];
  o[1] = R[3]*x + R[4]*y + R[5]*z + R[10];
  o[2] = R[6]*x + R[7]*y + R[8]*z + R[11];
}

// qn[n][h] = sum_p |qp_g|^2 ; kn[n][h] = sum_{p<4} |kp_g|^2
__global__ void point_norms(const float* qp_g, const float* kvp_g,
                            float* qn, float* kn)
{
  int idx = blockIdx.x * blockDim.x + threadIdx.x;
  if (idx >= NRES * NH) return;
  int n = idx / NH, h = idx % NH;
  float s = 0.f;
  for (int p = 0; p < PQ; p++) {
    const float* q = qp_g + (size_t)n * (NH*PQ*3) + h * (PQ*3) + p * 3;
    s += q[0]*q[0] + q[1]*q[1] + q[2]*q[2];
  }
  qn[idx] = s;
  s = 0.f;
  for (int p = 0; p < PQ; p++) {
    const float* k = kvp_g + (size_t)n * (NH*12*3) + h * 36 + p * 3;
    s += k[0]*k[0] + k[1]*k[1] + k[2]*k[2];
  }
  kn[idx] = s;
}

// ---------------------------------------------------------------------------
// IPA logits: one K=32 WMMA per 16x16 (i,j) tile per head.
// features: [c1*q (16) | c2h*qp_g (12) | 0..]  x  [k (16) | kp_g (12) | 0..]
// norm/bias/mask terms added exactly in f32 in the epilogue.
// grid (4, 32, 12), block 256 (8 waves, one j-tile each)
// ---------------------------------------------------------------------------
__global__ __launch_bounds__(256)
void ipa_logits(const float* __restrict__ q, const float* __restrict__ kv,
                const float* __restrict__ qp_g, const float* __restrict__ kvp_g,
                const float* __restrict__ qn, const float* __restrict__ kn,
                const float* __restrict__ bbias, const float* __restrict__ head_w,
                const float* __restrict__ mask, float* __restrict__ attn)
{
  const int h = blockIdx.z;
  const int i0 = blockIdx.y * 16;
  const int w = threadIdx.x >> 5, l = threadIdx.x & 31;
  const int j0 = (blockIdx.x * 8 + w) * 16;
  const int lm = l & 15, half = (l >> 4) & 1;

  const float hw = log1pf(expf(head_w[h]));   // softplus
  const float c1 = WLC * 0.25f;               // wL / sqrt(C_IPA)
  const float c2 = WLC * WCC * hw;

  FragAB a, b;
  const int i = i0 + lm;
#pragma unroll
  for (int e = 0; e < 16; e++) {
    int k = a_k_of(e, half);
    float v;
    if (k < 16)      v = c1 * q[(size_t)i * (NH*CI) + h * CI + k];
    else if (k < 28) v = c2 * qp_g[(size_t)i * (NH*PQ*3) + h * (PQ*3) + (k - 16)];
    else             v = 0.f;
    a.u[e] = f2bf(v);
  }
  const int j = j0 + lm;
#pragma unroll
  for (int e = 0; e < 16; e++) {
    int k = half * 16 + e;
    float v;
    if (k < 16)      v = kv[(size_t)j * (NH*2*CI) + h * (2*CI) + k];
    else if (k < 28) v = kvp_g[(size_t)j * (NH*12*3) + h * 36 + (k - 16)];
    else             v = 0.f;
    b.u[e] = f2bf(v);
  }
  FragCD acc;
  for (int r = 0; r < 8; r++) acc.f[r] = 0.f;
  acc.v = wmma_bf16(a, b, acc.v);

#pragma unroll
  for (int r = 0; r < 8; r++) {
    int ii = i0 + r + 8 * half;
    int jj = j0 + lm;
    float v = acc.f[r]
            - 0.5f * c2 * (qn[ii * NH + h] + kn[jj * NH + h])
            + WLC * bbias[((size_t)ii * NRES + jj) * NH + h]
            + 100000.f * (mask[ii] * mask[jj] - 1.f);
    attn[((size_t)h * NRES + ii) * NRES + jj] = v;
  }
}

// Softmax over last dim (512) — one wave32 per row.
__global__ void softmax_rows(float* attn)
{
  int wid = threadIdx.x >> 5, lane = threadIdx.x & 31;
  int row = blockIdx.x * 8 + wid;
  float* r = attn + (size_t)row * NRES;
  float vals[16];
  float mx = -1e30f;
#pragma unroll
  for (int t = 0; t < 16; t++) { vals[t] = r[lane + t * 32]; mx = fmaxf(mx, vals[t]); }
  for (int m = 16; m > 0; m >>= 1) mx = fmaxf(mx, __shfl_xor(mx, m, 32));
  float sum = 0.f;
#pragma unroll
  for (int t = 0; t < 16; t++) { vals[t] = expf(vals[t] - mx); sum += vals[t]; }
  for (int m = 16; m > 0; m >>= 1) sum += __shfl_xor(sum, m, 32);
  float inv = 1.f / sum;
#pragma unroll
  for (int t = 0; t < 16; t++) r[lane + t * 32] = vals[t] * inv;
}

// ---------------------------------------------------------------------------
// o = a@v (16 cols) and opt = a@vp (24 cols), K=512. grid (32, 12), block 96.
// ---------------------------------------------------------------------------
__global__ void attn_ov(const float* __restrict__ attn, const float* __restrict__ kv,
                        const float* __restrict__ kvp_g,
                        float* __restrict__ feat, float* __restrict__ optg)
{
  const int h = blockIdx.y;
  const int i0 = blockIdx.x * 16;
  const int w = threadIdx.x >> 5, l = threadIdx.x & 31;
  const int lm = l & 15, half = (l >> 4) & 1;
  const int c = w * 16 + lm;

  FragCD acc;
  for (int r = 0; r < 8; r++) acc.f[r] = 0.f;

  for (int kt = 0; kt < 16; kt++) {
    FragAB a, b;
#pragma unroll
    for (int e = 0; e < 16; e++) {
      int k = kt * 32 + a_k_of(e, half);
      a.u[e] = f2bf(attn[((size_t)h * NRES + (i0 + lm)) * NRES + k]);
    }
#pragma unroll
    for (int e = 0; e < 16; e++) {
      int j = kt * 32 + half * 16 + e;
      float v;
      if (c < 16) v = kv[(size_t)j * (NH*2*CI) + h * (2*CI) + 16 + c];
      else if (c < 40) {
        int cc = c - 16;
        v = kvp_g[(size_t)j * (NH*12*3) + h * 36 + (PQ + cc / 3) * 3 + cc % 3];
      } else v = 0.f;
      b.u[e] = f2bf(v);
    }
    acc.v = wmma_bf16(a, b, acc.v);
  }
#pragma unroll
  for (int r = 0; r < 8; r++) {
    int i = i0 + r + 8 * half;
    if (c < 16)      feat[(size_t)i * FEAT + h * CI + c] = acc.f[r];
    else if (c < 40) optg[(size_t)i * (NH*PV*3) + h * (PV*3) + (c - 16)] = acc.f[r];
  }
}

// ---------------------------------------------------------------------------
// opair[i][h][c] = sum_j a[h,i,j] * z_ln[i,j,c].  One block per residue i.
// M=16 (12 heads used), K=512, N=128. Block 256 (8 waves, one c-tile each).
// ---------------------------------------------------------------------------
__global__ __launch_bounds__(256)
void attn_opair(const float* __restrict__ attn, const bf16_t* __restrict__ z_ln,
                float* __restrict__ feat)
{
  const int i = blockIdx.x;
  const int w = threadIdx.x >> 5, l = threadIdx.x & 31;
  const int lm = l & 15, half = (l >> 4) & 1;
  const int c = w * 16 + lm;

  FragCD acc;
  for (int r = 0; r < 8; r++) acc.f[r] = 0.f;

  for (int kt = 0; kt < 16; kt++) {
    FragAB a, b;
#pragma unroll
    for (int e = 0; e < 16; e++) {
      int k = kt * 32 + a_k_of(e, half);
      a.u[e] = (lm < NH) ? f2bf(attn[((size_t)lm * NRES + i) * NRES + k]) : (bf16_t)0;
    }
#pragma unroll
    for (int e = 0; e < 16; e++) {
      int j = kt * 32 + half * 16 + e;
      b.u[e] = z_ln[((size_t)i * NRES + j) * CZ + c];
    }
    acc.v = wmma_bf16(a, b, acc.v);
  }
#pragma unroll
  for (int r = 0; r < 8; r++) {
    int h = r + 8 * half;
    if (h < NH)
      feat[(size_t)i * FEAT + (NH*CI + NH*PV*3 + NH*PV) + h * CZ + c] = acc.f[r];
  }
}

// opt_loc = R^T (opt - t); also norm. thread per (n,h,p).
__global__ void opt_transform(const float* optg, const float* Rt, float* feat)
{
  int idx = blockIdx.x * blockDim.x + threadIdx.x;
  if (idx >= NRES * NH * PV) return;
  int n = idx / (NH * PV);
  int hp = idx % (NH * PV);
  const float* R = Rt + n * 12;
  const float* og = optg + (size_t)n * (NH*PV*3) + hp * 3;
  float dx = og[0] - R[9], dy = og[1] - R[10], dz = og[2] - R[11];
  float l0 = R[0]*dx + R[3]*dy + R[6]*dz;
  float l1 = R[1]*dx + R[4]*dy + R[7]*dz;
  float l2 = R[2]*dx + R[5]*dy + R[8]*dz;
  float* f = feat + (size_t)n * FEAT;
  f[NH*CI + hp * 3 + 0] = l0;
  f[NH*CI + hp * 3 + 1] = l1;
  f[NH*CI + hp * 3 + 2] = l2;
  f[NH*CI + NH*PV*3 + hp] = sqrtf(l0*l0 + l1*l1 + l2*l2 + 1e-8f);
}

// angle normalization: ang = un / sqrt(si^2+co^2+EPS)
__global__ void norm_angles(const float* un, float* ang)
{
  int idx = blockIdx.x * blockDim.x + threadIdx.x;
  if (idx >= NRES * 9) return;
  float si = un[idx * 2 + 0], co = un[idx * 2 + 1];
  float inv = rsqrtf(si * si + co * co + 1e-8f);
  ang[idx * 2 + 0] = si * inv;
  ang[idx * 2 + 1] = co * inv;
}

// ---------------------------------------------------------------------------
// Torsion frames + atom placement (final block only). thread per (n, out-atom)
// ---------------------------------------------------------------------------
__global__ void place_atoms(const int* rttype, const float* ang,
                            const float* R1, const float* R2,
                            const float* dframes, const int* group_idx,
                            const float* atom_mask, const float* lit_pos,
                            float* out)
{
  int idx = blockIdx.x * blockDim.x + threadIdx.x;
  if (idx >= NRES * 23) return;
  int n = idx / 23, a = idx % 23;
  int sa = (a == 3) ? 4 : ((a == 4) ? 3 : a);   // swap permutation
  int rt = rttype[n];
  int g = group_idx[rt * 23 + sa];
  const int sel[9] = {0, 0, 0, 1, 1, 0, 0, 0, 1};
  const float* df = dframes + rt * 144 + g * 16;   // 4x4
  float si = ang[n * 18 + g * 2 + 0];
  float co = ang[n * 18 + g * 2 + 1];
  const float* lit = lit_pos + rt * 69 + sa * 3;
  float v[3];
#pragma unroll
  for (int i = 0; i < 3; i++) {
    float r0 = df[i*4+0];
    float r1 = df[i*4+1] * co + df[i*4+2] * si;
    float r2 = -df[i*4+1] * si + df[i*4+2] * co;
    v[i] = r0 * lit[0] + r1 * lit[1] + r2 * lit[2] + df[i*4+3];
  }
  const float* F = (sel[g] ? R2 : R1) + n * 12;
  float m = atom_mask[rt * 23 + sa];
#pragma unroll
  for (int i = 0; i < 3; i++)
    out[n * 69 + a * 3 + i] =
        (F[i*3+0] * v[0] + F[i*3+1] * v[1] + F[i*3+2] * v[2] + F[9 + i]) * m;
}

// ---------------------------------------------------------------------------
// Host orchestration
// ---------------------------------------------------------------------------
static inline void gemmF(hipStream_t st, const float* A, const bf16_t* Wt,
                         const float* b, const float* resid, float* out,
                         int M, int K, int N, int flags)
{
  dim3 g((N + 63) / 64, (M + 63) / 64);
  gemm_wmma<float><<<g, 128, 0, st>>>(A, Wt, b, resid, out, M, K, N, flags);
}
static inline void gemmB(hipStream_t st, const bf16_t* A, const bf16_t* Wt,
                         const float* b, const float* resid, float* out,
                         int M, int K, int N, int flags)
{
  dim3 g((N + 63) / 64, (M + 63) / 64);
  gemm_wmma<bf16_t><<<g, 128, 0, st>>>(A, Wt, b, resid, out, M, K, N, flags);
}

extern "C" void kernel_launch(void* const* d_in, const int* in_sizes, int n_in,
                              void* d_out, int out_size, void* d_ws, size_t ws_size,
                              hipStream_t stream)
{
  (void)in_sizes; (void)n_in; (void)out_size; (void)ws_size;
  const float* single    = (const float*)d_in[0];
  const float* pair      = (const float*)d_in[1];
  const int*   rttype    = (const int*)  d_in[2];
  const float* mask      = (const float*)d_in[3];
  const float* ln_s_g    = (const float*)d_in[4];
  const float* ln_s_b    = (const float*)d_in[5];
  const float* ln_z_g    = (const float*)d_in[6];
  const float* ln_z_b    = (const float*)d_in[7];
  const float* lin_in_w  = (const float*)d_in[8];
  const float* lin_in_b  = (const float*)d_in[9];
  const float* q_w       = (const float*)d_in[10];
  const float* q_b       = (const float*)d_in[11];
  const float* kv_w      = (const float*)d_in[12];
  const float* kv_b      = (const float*)d_in[13];
  const float* qp_w      = (const float*)d_in[14];
  const float* qp_b      = (const float*)d_in[15];
  const float* kvp_w     = (const float*)d_in[16];
  const float* kvp_b     = (const float*)d_in[17];
  const float* zb_w      = (const float*)d_in[18];
  const float* zb_b      = (const float*)d_in[19];
  const float* head_w    = (const float*)d_in[20];
  const float* ipa_out_w = (const float*)d_in[21];
  const float* ipa_out_b = (const float*)d_in[22];
  const float* ln_ipa_g  = (const float*)d_in[23];
  const float* ln_ipa_b  = (const float*)d_in[24];
  const float* t_1w      = (const float*)d_in[25];
  const float* t_1b      = (const float*)d_in[26];
  const float* t_2w      = (const float*)d_in[27];
  const float* t_2b      = (const float*)d_in[28];
  const float* t_3w      = (const float*)d_in[29];
  const float* t_3b      = (const float*)d_in[30];
  const float* t_ln_g    = (const float*)d_in[31];
  const float* t_ln_b    = (const float*)d_in[32];
  const float* ref_w     = (const float*)d_in[33];
  const float* ref_b     = (const float*)d_in[34];
  const float* ar_in_w   = (const float*)d_in[35];
  const float* ar_in_b   = (const float*)d_in[36];
  const float* ar_init_w = (const float*)d_in[37];
  const float* ar_init_b = (const float*)d_in[38];
  const float* ar_w1[2]  = {(const float*)d_in[39], (const float*)d_in[43]};
  const float* ar_b1[2]  = {(const float*)d_in[40], (const float*)d_in[44]};
  const float* ar_w2[2]  = {(const float*)d_in[41], (const float*)d_in[45]};
  const float* ar_b2[2]  = {(const float*)d_in[42], (const float*)d_in[46]};
  const float* ar_out_w  = (const float*)d_in[47];
  const float* ar_out_b  = (const float*)d_in[48];
  const float* dframes   = (const float*)d_in[49];
  const int*   group_idx = (const int*)  d_in[50];
  const float* atom_mask = (const float*)d_in[51];
  const float* lit_pos   = (const float*)d_in[52];

  // workspace carving
  char* wp = (char*)d_ws;
  auto alloc = [&](size_t bytes) -> void* {
    void* p = (void*)wp;
    wp += (bytes + 255) & ~(size_t)255;
    return p;
  };
  const size_t NN = (size_t)NRES * NRES;
  float*  s      = (float*) alloc((size_t)NRES * CS * 4);
  float*  s_init = (float*) alloc((size_t)NRES * CS * 4);
  bf16_t* z_ln   = (bf16_t*)alloc(NN * CZ * 2);
  float*  bbias  = (float*) alloc(NN * NH * 4);
  float*  attn   = (float*) alloc((size_t)NH * NN * 4);
  float*  qb     = (float*) alloc((size_t)NRES * NH * CI * 4);
  float*  kvb    = (float*) alloc((size_t)NRES * NH * 2 * CI * 4);
  float*  qp_l   = (float*) alloc((size_t)NRES * NH * PQ * 3 * 4);
  float*  kvp_l  = (float*) alloc((size_t)NRES * NH * 12 * 3 * 4);
  float*  qp_g   = (float*) alloc((size_t)NRES * NH * PQ * 3 * 4);
  float*  kvp_g  = (float*) alloc((size_t)NRES * NH * 12 * 3 * 4);
  float*  qn     = (float*) alloc((size_t)NRES * NH * 4);
  float*  kn     = (float*) alloc((size_t)NRES * NH * 4);
  float*  optg   = (float*) alloc((size_t)NRES * NH * PV * 3 * 4);
  float*  feat   = (float*) alloc((size_t)NRES * FEAT * 4);
  float*  h1     = (float*) alloc((size_t)NRES * CS * 4);
  float*  h2     = (float*) alloc((size_t)NRES * CS * 4);
  float*  xyz    = (float*) alloc((size_t)NRES * 15 * 4);
  float*  dref   = (float*) alloc((size_t)NRES * 15 * 4);
  float*  Rf     = (float*) alloc((size_t)NRES * 12 * 4);
  float*  R1     = (float*) alloc((size_t)NRES * 12 * 4);
  float*  R2     = (float*) alloc((size_t)NRES * 12 * 4);
  float*  ar_a   = (float*) alloc((size_t)NRES * 128 * 4);
  float*  ar_h   = (float*) alloc((size_t)NRES * 128 * 4);
  float*  un     = (float*) alloc((size_t)NRES * 18 * 4);
  float*  ang    = (float*) alloc((size_t)NRES * 18 * 4);

  // packed bf16 weights, transposed [Npad][K], zero-padded
  auto packW = [&](const float* W, int K, int N) -> bf16_t* {
    int Npad = (N + 63) & ~63;
    bf16_t* p = (bf16_t*)alloc((size_t)Npad * K * 2);
    int tot = Npad * K;
    pack_weight<<<(tot + 255) / 256, 256, 0, stream>>>(W, p, K, N, Npad);
    return p;
  };
  bf16_t* Wlin   = packW(lin_in_w,  CS, CS);
  bf16_t* Wq     = packW(q_w,       CS, NH*CI);
  bf16_t* Wkv    = packW(kv_w,      CS, NH*2*CI);
  bf16_t* Wqp    = packW(qp_w,      CS, NH*PQ*3);
  bf16_t* Wkvp   = packW(kvp_w,     CS, NH*12*3);
  bf16_t* Wzb    = packW(zb_w,      CZ, NH);
  bf16_t* Wipa   = packW(ipa_out_w, FEAT, CS);
  bf16_t* Wt1    = packW(t_1w,      CS, CS);
  bf16_t* Wt2    = packW(t_2w,      CS, CS);
  bf16_t* Wt3    = packW(t_3w,      CS, CS);
  bf16_t* Wref   = packW(ref_w,     CS, 15);
  bf16_t* Warin  = packW(ar_in_w,   CS, 128);
  bf16_t* Warini = packW(ar_init_w, CS, 128);
  bf16_t* War1[2] = {packW(ar_w1[0], 128, 128), packW(ar_w1[1], 128, 128)};
  bf16_t* War2[2] = {packW(ar_w2[0], 128, 128), packW(ar_w2[1], 128, 128)};
  bf16_t* Warout = packW(ar_out_w,  128, 18);

  // ---- setup (once) ----
  ln_f32<<<NRES, 128, 0, stream>>>(single, ln_s_g, ln_s_b, s_init, CS);
  gemmF(stream, s_init, Wlin, lin_in_b, nullptr, s, NRES, CS, CS, 0);
  ln_f32_to_bf16<<<(unsigned)NN, 128, 0, stream>>>(pair, ln_z_g, ln_z_b, z_ln, CZ);
  gemmB(stream, z_ln, Wzb, zb_b, nullptr, bbias, (int)NN, CZ, NH, 0);
  fill0<<<30, 256, 0, stream>>>(xyz, NRES * 15);

  // ---- 8 structure blocks ----
  for (int blk = 0; blk < 8; blk++) {
    frames_from3<<<2, 256, 0, stream>>>(xyz, 0, 1, 2, 1.f, Rf);
    gemmF(stream, s, Wq,   q_b,   nullptr, qb,    NRES, CS, NH*CI,   0);
    gemmF(stream, s, Wkv,  kv_b,  nullptr, kvb,   NRES, CS, NH*2*CI, 0);
    gemmF(stream, s, Wqp,  qp_b,  nullptr, qp_l,  NRES, CS, NH*PQ*3, 0);
    gemmF(stream, s, Wkvp, kvp_b, nullptr, kvp_l, NRES, CS, NH*12*3, 0);
    to_global<<<(NRES*NH*PQ + 255) / 256, 256, 0, stream>>>(qp_l,  Rf, qp_g,  NH*PQ);
    to_global<<<(NRES*NH*12 + 255) / 256, 256, 0, stream>>>(kvp_l, Rf, kvp_g, NH*12);
    point_norms<<<(NRES*NH + 255) / 256, 256, 0, stream>>>(qp_g, kvp_g, qn, kn);
    ipa_logits<<<dim3(4, 32, NH), 256, 0, stream>>>(qb, kvb, qp_g, kvp_g, qn, kn,
                                                    bbias, head_w, mask, attn);
    softmax_rows<<<NH * NRES / 8, 256, 0, stream>>>(attn);
    attn_ov<<<dim3(32, NH), 96, 0, stream>>>(attn, kvb, kvp_g, feat, optg);
    attn_opair<<<NRES, 256, 0, stream>>>(attn, z_ln, feat);
    opt_transform<<<(NRES*NH*PV + 255) / 256, 256, 0, stream>>>(optg, Rf, feat);
    gemmF(stream, feat, Wipa, ipa_out_b, s, s, NRES, FEAT, CS, GT_RESID);
    ln_f32<<<NRES, 128, 0, stream>>>(s, ln_ipa_g, ln_ipa_b, s, CS);
    gemmF(stream, s,  Wt1, t_1b, nullptr, h1, NRES, CS, CS, GT_RELU_OUT);
    gemmF(stream, h1, Wt2, t_2b, nullptr, h2, NRES, CS, CS, GT_RELU_OUT);
    gemmF(stream, h2, Wt3, t_3b, s, s, NRES, CS, CS, GT_RESID);
    ln_f32<<<NRES, 128, 0, stream>>>(s, t_ln_g, t_ln_b, s, CS);
    gemmF(stream, s, Wref, ref_b, nullptr, dref, NRES, CS, 15, 0);
    addvec<<<30, 256, 0, stream>>>(xyz, dref, NRES * 15);
  }

  // ---- angle resnet (only affects pred; run once after the loop) ----
  gemmF(stream, s,      Warin,  ar_in_b,   nullptr, ar_a, NRES, CS, 128, GT_RELU_IN);
  gemmF(stream, s_init, Warini, ar_init_b, ar_a,    ar_a, NRES, CS, 128, GT_RELU_IN | GT_RESID);
  for (int i = 0; i < 2; i++) {
    gemmF(stream, ar_a, War1[i], ar_b1[i], nullptr, ar_h, NRES, 128, 128, GT_RELU_IN);
    gemmF(stream, ar_h, War2[i], ar_b2[i], ar_a,    ar_a, NRES, 128, 128, GT_RELU_IN | GT_RESID);
  }
  gemmF(stream, ar_a, Warout, ar_out_b, nullptr, un, NRES, 128, 18, GT_RELU_IN);
  norm_angles<<<(NRES*9 + 255) / 256, 256, 0, stream>>>(un, ang);

  // ---- final frames + atom placement ----
  frames_from3<<<2, 256, 0, stream>>>(xyz, 0, 1, 2, 10.f, R1);
  frames_from3<<<2, 256, 0, stream>>>(xyz, 4, 3, 2, 10.f, R2);
  place_atoms<<<(NRES*23 + 255) / 256, 256, 0, stream>>>(
      rttype, ang, R1, R2, dframes, group_idx, atom_mask, lit_pos, (float*)d_out);
}
